// FNO3d_5841155523236
// MI455X (gfx1250) — matrix operand order repaired and checked by reading, not compiled, not run on hardware
//
#include <hip/hip_runtime.h>
#include <hip/hip_fp16.h>

// ---------------------------------------------------------------------------
// FNO3d forward for MI455X (gfx1250, wave32, WMMA).
// Layout: activations channel-last  [b][x][y][z][c=32]  (GEMM rows contiguous)
// Spectral temporaries:             [....][kz][re/im][c]
// Channel-mix GEMMs use v_wmma_f32_16x16x32_f16 (f16 in, f32 accum); B-matrix
// fragments and biases are hoisted into registers and reused over 4 M-tiles
// per wave. DFT stages are f32 VALU with LDS twiddle tables.
// ---------------------------------------------------------------------------

constexpr int BATCH = 4;
constexpr int NX = 64, NY = 64;
constexpr int NZ0 = 40;          // pre-pad z
constexpr int NZ  = 46;          // padded z
constexpr int CH  = 32;
constexpr int MK  = 8;           // z modes
constexpr int MXY = 16;          // x/y modes incl. both corners (0..7, 56..63)
constexpr int NCOL  = BATCH * NX * NY;          // 16384
constexpr int NPIX  = NCOL * NZ;                // 753664
constexpr int NPIX0 = NCOL * NZ0;               // 655360
constexpr int T_PW  = 4;         // M-tiles per wave, pointwise kernels
constexpr int T_Q   = 4;         // M-tiles per wave, projection head

typedef __attribute__((ext_vector_type(16))) _Float16 v16h;
typedef __attribute__((ext_vector_type(8)))  float    v8f;

__device__ __forceinline__ float gelu_f(float x) {
    return 0.5f * x * (1.0f + erff(x * 0.70710678118654752f));
}
__device__ __forceinline__ float sigm_f(float x) {
    return 1.0f / (1.0f + expf(-x));
}

// ---- WMMA fragment helpers (wave32 layouts per CDNA5 ISA 7.12.2) ----------
// A (16x32 f16): lane L holds row M=L&15; lanes<16 hold K{0-7,16-23},
// lanes>=16 hold K{8-15,24-31}; VGPR pairs are consecutive K.
__device__ __forceinline__ v16h load_a_f32(const float* rowptr, int lane) {
    const int k0 = (lane < 16) ? 0 : 8;
    v16h a;
#pragma unroll
    for (int j = 0; j < 8; ++j) a[j]     = (_Float16)rowptr[k0 + j];
#pragma unroll
    for (int j = 0; j < 8; ++j) a[8 + j] = (_Float16)rowptr[16 + k0 + j];
    return a;
}
__device__ __forceinline__ v16h load_a_f16(const _Float16* rowptr, int lane) {
    const int k0 = (lane < 16) ? 0 : 8;
    v16h a;
#pragma unroll
    for (int j = 0; j < 8; ++j) a[j]     = rowptr[k0 + j];
#pragma unroll
    for (int j = 0; j < 8; ++j) a[8 + j] = rowptr[16 + k0 + j];
    return a;
}
// B (32x16 f16): lane L holds column N=L&15; lanes<16 K{0-7,16-23}, lanes>=16 K{8-15,24-31}.
// W stored [o][i] f16 row-major with i-stride 32, so column N == row o of W.
__device__ __forceinline__ v16h load_b_f16(const _Float16* W, int ntile, int lane) {
    const int n  = ntile * 16 + (lane & 15);
    const int k0 = (lane < 16) ? 0 : 8;
    const _Float16* r = W + n * 32 + k0;
    v16h b;
#pragma unroll
    for (int j = 0; j < 8; ++j) { b[j] = r[j]; b[8 + j] = r[16 + j]; }
    return b;
}
__device__ __forceinline__ v8f wmma32(v16h a, v16h b, v8f c) {
    return __builtin_amdgcn_wmma_f32_16x16x32_f16(false, a, false, b, (short)0, c,
                                                  false, false);
}

// ---------------------------------------------------------------------------
// Weight f32 -> f16 conversion
__global__ void k_cvt(const float* __restrict__ src, _Float16* __restrict__ dst, int n) {
    int i = blockIdx.x * 256 + threadIdx.x;
    if (i < n) dst[i] = (_Float16)src[i];
}

// ---------------------------------------------------------------------------
// Lift: concat(x, grid) @ p_w^T + p_b, channel-first->channel-last, z-pad zeros.
__global__ __launch_bounds__(256) void k_lift(const float* __restrict__ X,
                                              const float* __restrict__ pw,
                                              const float* __restrict__ pb,
                                              float* __restrict__ out) {
    __shared__ float spw[320];
    __shared__ float spb[32];
    if (threadIdx.x < 320) spw[threadIdx.x] = pw[threadIdx.x];
    if (threadIdx.x < 32)  spb[threadIdx.x] = pb[threadIdx.x];
    __syncthreads();
    const int col = blockIdx.x;                 // ((b*64+xx)*64+yy)
    const int yy = col & 63;
    const int xx = (col >> 6) & 63;
    const float gx = xx * (1.0f / 63.0f);
    const float gy = yy * (1.0f / 63.0f);
    const int c  = threadIdx.x & 31;
    const int zg = threadIdx.x >> 5;
    for (int zi = 0; zi < 6; ++zi) {
        const int z = zi * 8 + zg;
        if (z >= NZ) continue;
        float v = 0.0f;
        if (z < NZ0) {
            const float* f = X + (col * NZ0 + z) * 7;
            const float gz = z * (1.0f / 39.0f);
            v = spb[c];
#pragma unroll
            for (int i = 0; i < 7; ++i) v += f[i] * spw[c * 10 + i];
            v += gx * spw[c * 10 + 7] + gy * spw[c * 10 + 8] + gz * spw[c * 10 + 9];
        }
        out[(col * NZ + z) * CH + c] = v;
    }
}

// ---------------------------------------------------------------------------
// Forward partial DFT over z: [col][z][c] -> [col][kz][ri][c]
__global__ __launch_bounds__(256) void k_dft_z(const float* __restrict__ in,
                                               float* __restrict__ out) {
    __shared__ float cw[MK * NZ];
    __shared__ float sw[MK * NZ];
    for (int t = threadIdx.x; t < MK * NZ; t += 256) {
        const int k = t / NZ, z = t - k * NZ;
        const int ph = (k * z) % NZ;
        float s, c;
        __sincosf(-6.2831853071795864f * (float)ph / (float)NZ, &s, &c);
        cw[t] = c; sw[t] = s;
    }
    __syncthreads();
    const int c  = threadIdx.x & 31;
    const int cl = threadIdx.x >> 5;
    const int col = blockIdx.x * 8 + cl;
    const float* src = in + col * NZ * CH + c;
    float re[MK], im[MK];
#pragma unroll
    for (int k = 0; k < MK; ++k) { re[k] = 0.0f; im[k] = 0.0f; }
    for (int z = 0; z < NZ; ++z) {
        const float v = src[z * CH];
#pragma unroll
        for (int k = 0; k < MK; ++k) {
            re[k] += v * cw[k * NZ + z];
            im[k] += v * sw[k * NZ + z];
        }
    }
    float* dst = out + col * MK * 2 * CH + c;
#pragma unroll
    for (int k = 0; k < MK; ++k) {
        dst[(k * 2 + 0) * CH] = re[k];
        dst[(k * 2 + 1) * CH] = im[k];
    }
}

// Forward DFT over y: [bx][y][kz][ri][c] -> [bx][ky][kz][ri][c]
__global__ __launch_bounds__(256) void k_dft_y(const float* __restrict__ in,
                                               float* __restrict__ out) {
    const int bi = blockIdx.x;                  // (b*64+xx)*16 + ky
    const int ky = bi & 15;
    const int bx = bi >> 4;
    const int f  = (ky < 8) ? ky : 48 + ky;
    __shared__ float cw[64];
    __shared__ float sw[64];
    if (threadIdx.x < 64) {
        const int ph = (f * threadIdx.x) & 63;
        float s, c;
        __sincosf(-6.2831853071795864f * (float)ph / 64.0f, &s, &c);
        cw[threadIdx.x] = c; sw[threadIdx.x] = s;
    }
    __syncthreads();
    const int c  = threadIdx.x & 31;
    const int kz = threadIdx.x >> 5;
    const float* src = in + ((bx * 64 * MK + kz) * 2) * CH + c;   // y=0
    float aR = 0.0f, aI = 0.0f;
    for (int y = 0; y < 64; ++y) {
        const float* p = src + y * (MK * 2 * CH);
        const float xr = p[0], xi = p[CH];
        aR += xr * cw[y] - xi * sw[y];
        aI += xi * cw[y] + xr * sw[y];
    }
    float* dst = out + (((bx * MXY + ky) * MK + kz) * 2) * CH + c;
    dst[0]  = aR;
    dst[CH] = aI;
}

// Forward DFT over x: [b][xx][ky][kz][ri][c] -> [b][kx][ky][kz][ri][c]
__global__ __launch_bounds__(256) void k_dft_x(const float* __restrict__ in,
                                               float* __restrict__ out) {
    const int bi = blockIdx.x;                  // (b*16+kx)*16 + ky
    const int ky = bi & 15;
    const int t2 = bi >> 4;
    const int kx = t2 & 15;
    const int b  = t2 >> 4;
    const int f  = (kx < 8) ? kx : 48 + kx;
    __shared__ float cw[64];
    __shared__ float sw[64];
    if (threadIdx.x < 64) {
        const int ph = (f * threadIdx.x) & 63;
        float s, c;
        __sincosf(-6.2831853071795864f * (float)ph / 64.0f, &s, &c);
        cw[threadIdx.x] = c; sw[threadIdx.x] = s;
    }
    __syncthreads();
    const int c  = threadIdx.x & 31;
    const int kz = threadIdx.x >> 5;
    const float* src = in + ((((b * 64) * MXY + ky) * MK + kz) * 2) * CH + c;  // xx=0
    const int xs = MXY * MK * 2 * CH;           // 8192
    float aR = 0.0f, aI = 0.0f;
    for (int xx = 0; xx < 64; ++xx) {
        const float xr = src[xx * xs], xi = src[xx * xs + CH];
        aR += xr * cw[xx] - xi * sw[xx];
        aI += xi * cw[xx] + xr * sw[xx];
    }
    float* dst = out + ((((b * MXY + kx) * MXY + ky) * MK + kz) * 2) * CH + c;
    dst[0]  = aR;
    dst[CH] = aI;
}

// Per-mode 32x32 complex channel mix (corner-selected weight).
__global__ __launch_bounds__(256) void k_modemul(const float* __restrict__ Sm,
                                                 const float* __restrict__ w1,
                                                 const float* __restrict__ w2,
                                                 const float* __restrict__ w3,
                                                 const float* __restrict__ w4,
                                                 float* __restrict__ Ym) {
    __shared__ float sx[512];                   // [kz][ri][i]
    const int bi = blockIdx.x;                  // (b*16+kx)*16+ky
    const int ky = bi & 15;
    const int kx = (bi >> 4) & 15;
    const int base = bi * 512;
    sx[threadIdx.x]       = Sm[base + threadIdx.x];
    sx[256 + threadIdx.x] = Sm[base + 256 + threadIdx.x];
    __syncthreads();
    const float* W = (kx < 8) ? ((ky < 8) ? w1 : w3) : ((ky < 8) ? w2 : w4);
    const int dx = kx & 7, dy = ky & 7;
    const int o  = threadIdx.x & 31;
    const int kz = threadIdx.x >> 5;
    // w idx: i*(32*8*8*8*2) + o*(8*8*8*2) + ((dx*8+dy)*8+kz)*2 + ri
    const float* wp = W + (size_t)o * 1024 + ((dx * 8 + dy) * 8 + kz) * 2;
    float aR = 0.0f, aI = 0.0f;
#pragma unroll 4
    for (int i = 0; i < 32; ++i) {
        const float xr = sx[(kz * 2 + 0) * 32 + i];
        const float xi = sx[(kz * 2 + 1) * 32 + i];
        const float wr = wp[(size_t)i * 32768];
        const float wi = wp[(size_t)i * 32768 + 1];
        aR += xr * wr - xi * wi;
        aI += xr * wi + xi * wr;
    }
    Ym[base + (kz * 2 + 0) * 32 + o] = aR;
    Ym[base + (kz * 2 + 1) * 32 + o] = aI;
}

// Inverse DFT over x modes: [b][kx][ky][kz][ri][oc] -> [b][xx][ky][kz][ri][oc] (scale 1/64)
__global__ __launch_bounds__(256) void k_idft_x(const float* __restrict__ Ym,
                                                float* __restrict__ out) {
    const int bi = blockIdx.x;                  // (b*64+xx)*16 + ky
    const int ky = bi & 15;
    const int t2 = bi >> 4;
    const int xx = t2 & 63;
    const int b  = t2 >> 6;
    __shared__ float cw[16];
    __shared__ float sw[16];
    if (threadIdx.x < 16) {
        const int j = threadIdx.x;
        const int f = (j < 8) ? j : 48 + j;
        const int ph = (f * xx) & 63;
        float s, c;
        __sincosf(6.2831853071795864f * (float)ph / 64.0f, &s, &c);
        cw[j] = c; sw[j] = s;
    }
    __syncthreads();
    const int oc = threadIdx.x & 31;
    const int kz = threadIdx.x >> 5;
    const float* src = Ym + (((b * MXY * MXY + ky) * MK + kz) * 2) * CH + oc;  // j=0
    const int js = MXY * MK * 2 * CH;           // 8192
    float aR = 0.0f, aI = 0.0f;
#pragma unroll
    for (int j = 0; j < 16; ++j) {
        const float xr = src[j * js], xi = src[j * js + CH];
        aR += xr * cw[j] - xi * sw[j];
        aI += xi * cw[j] + xr * sw[j];
    }
    const float sc = 1.0f / 64.0f;
    float* dst = out + ((bi * MK + kz) * 2) * CH + oc;
    dst[0]  = aR * sc;
    dst[CH] = aI * sc;
}

// Inverse DFT over y modes: [bx][ky][kz][ri][oc] -> [col][kz][ri][oc] (scale 1/64)
__global__ __launch_bounds__(256) void k_idft_y(const float* __restrict__ Yx,
                                                float* __restrict__ out) {
    const int col = blockIdx.x;                 // (b*64+xx)*64 + yy
    const int yy = col & 63;
    const int bx = col >> 6;
    __shared__ float cw[16];
    __shared__ float sw[16];
    if (threadIdx.x < 16) {
        const int j = threadIdx.x;
        const int f = (j < 8) ? j : 48 + j;
        const int ph = (f * yy) & 63;
        float s, c;
        __sincosf(6.2831853071795864f * (float)ph / 64.0f, &s, &c);
        cw[j] = c; sw[j] = s;
    }
    __syncthreads();
    const int oc = threadIdx.x & 31;
    const int kz = threadIdx.x >> 5;
    const float* src = Yx + ((bx * MXY * MK + kz) * 2) * CH + oc;   // ky=0
    const int js = MK * 2 * CH;                 // 512
    float aR = 0.0f, aI = 0.0f;
#pragma unroll
    for (int j = 0; j < 16; ++j) {
        const float xr = src[j * js], xi = src[j * js + CH];
        aR += xr * cw[j] - xi * sw[j];
        aI += xi * cw[j] + xr * sw[j];
    }
    const float sc = 1.0f / 64.0f;
    float* dst = out + ((col * MK + kz) * 2) * CH + oc;
    dst[0]  = aR * sc;
    dst[CH] = aI * sc;
}

// Inverse rfft over z (Hermitian, modes 0..7 of 24): -> real [col][z][oc], scale 1/46.
__global__ __launch_bounds__(256) void k_idft_z(const float* __restrict__ Yxy,
                                                float* __restrict__ out) {
    __shared__ float sIn[512];                  // [kz][ri][oc]
    __shared__ float cw[MK * NZ];
    __shared__ float sw[MK * NZ];
    const int col = blockIdx.x;
    for (int t = threadIdx.x; t < MK * NZ; t += 256) {
        const int k = t / NZ, z = t - k * NZ;
        const int ph = (k * z) % NZ;
        float s, c;
        __sincosf(6.2831853071795864f * (float)ph / (float)NZ, &s, &c);
        cw[t] = c; sw[t] = s;
    }
    sIn[threadIdx.x]       = Yxy[col * 512 + threadIdx.x];
    sIn[256 + threadIdx.x] = Yxy[col * 512 + 256 + threadIdx.x];
    __syncthreads();
    const int oc = threadIdx.x & 31;
    const int zg = threadIdx.x >> 5;
    const float inv = 1.0f / (float)NZ;
    for (int zi = 0; zi < 6; ++zi) {
        const int z = zi * 8 + zg;
        if (z >= NZ) continue;
        float acc = sIn[oc];                    // k=0 real part (imag discarded, irfft semantics)
#pragma unroll
        for (int k = 1; k < MK; ++k) {
            const float re = sIn[(k * 2 + 0) * 32 + oc];
            const float im = sIn[(k * 2 + 1) * 32 + oc];
            acc += 2.0f * (re * cw[k * NZ + z] - im * sw[k * NZ + z]);
        }
        out[(col * NZ + z) * CH + oc] = acc * inv;
    }
}

// ---------------------------------------------------------------------------
// Fused per-block pointwise update, B fragments hoisted, T_PW tiles per wave:
//   x1buf <- act( gelu(x1*W1 + b1)*W2 + b2 + x*Ws + bs )
__global__ __launch_bounds__(128) void k_block_pw(float* __restrict__ x1buf,
                                                  const float* __restrict__ xbuf,
                                                  const _Float16* __restrict__ W1,
                                                  const float* __restrict__ b1,
                                                  const _Float16* __restrict__ W2,
                                                  const float* __restrict__ b2,
                                                  const _Float16* __restrict__ Ws,
                                                  const float* __restrict__ bs,
                                                  int act) {
    __shared__ _Float16 sh[4][16][32];          // per-wave hidden tile
    const int l = threadIdx.x & 31;
    const int w = threadIdx.x >> 5;
    const int tile0 = (blockIdx.x * 4 + w) * T_PW;
    const int m0 = l & 15;

    // Hoist weight fragments + biases (reused across T_PW tiles).
    const v16h B10 = load_b_f16(W1, 0, l), B11 = load_b_f16(W1, 1, l);
    const v16h B20 = load_b_f16(W2, 0, l), B21 = load_b_f16(W2, 1, l);
    const v16h BS0 = load_b_f16(Ws, 0, l), BS1 = load_b_f16(Ws, 1, l);
    const float b1lo = b1[m0], b1hi = b1[16 + m0];
    const float bolo = b2[m0] + bs[m0], bohi = b2[16 + m0] + bs[16 + m0];

    for (int t = 0; t < T_PW; ++t) {
        const int p = (tile0 + t) * 16;
        __builtin_prefetch(xbuf + (p + m0) * CH, 0, 3);

        // GEMM 1: hidden = gelu(x1 * W1 + b1)
        v16h a1 = load_a_f32(x1buf + (p + m0) * CH, l);
        v8f h0 = {}, h1 = {};
        h0 = wmma32(a1, B10, h0);
        h1 = wmma32(a1, B11, h1);
#pragma unroll
        for (int r = 0; r < 8; ++r) {
            const int m = (l < 16) ? r : r + 8;
            sh[w][m][m0]      = (_Float16)gelu_f(h0[r] + b1lo);
            sh[w][m][16 + m0] = (_Float16)gelu_f(h1[r] + b1hi);
        }
        // GEMM 2 + GEMM 3 (skip), chained into the same accumulators.
        v16h ah = load_a_f16(&sh[w][m0][0], l); // wave-internal LDS, DS in-order
        v8f o0 = {}, o1 = {};
        o0 = wmma32(ah, B20, o0);
        o1 = wmma32(ah, B21, o1);
        v16h ax = load_a_f32(xbuf + (p + m0) * CH, l);
        o0 = wmma32(ax, BS0, o0);
        o1 = wmma32(ax, BS1, o1);
#pragma unroll
        for (int r = 0; r < 8; ++r) {
            const int m = (l < 16) ? r : r + 8;
            float v0 = o0[r] + bolo;
            float v1 = o1[r] + bohi;
            if (act == 1) { v0 = gelu_f(v0); v1 = gelu_f(v1); }
            else          { v0 = sigm_f(v0); v1 = sigm_f(v1); }
            x1buf[(p + m) * CH + m0]      = v0;
            x1buf[(p + m) * CH + 16 + m0] = v1;
        }
    }
}

// Block 2 (spectral branch is dead in the reference): out = gelu(x + x*Ws + bs)
__global__ __launch_bounds__(128) void k_skip_only(const float* __restrict__ xin,
                                                   const _Float16* __restrict__ Ws,
                                                   const float* __restrict__ bs,
                                                   float* __restrict__ xout) {
    const int l = threadIdx.x & 31;
    const int w = threadIdx.x >> 5;
    const int tile0 = (blockIdx.x * 4 + w) * T_PW;
    const int m0 = l & 15;
    const v16h BS0 = load_b_f16(Ws, 0, l), BS1 = load_b_f16(Ws, 1, l);
    const float bslo = bs[m0], bshi = bs[16 + m0];
    for (int t = 0; t < T_PW; ++t) {
        const int p = (tile0 + t) * 16;
        v16h ax = load_a_f32(xin + (p + m0) * CH, l);
        v8f o0 = {}, o1 = {};
        o0 = wmma32(ax, BS0, o0);
        o1 = wmma32(ax, BS1, o1);
#pragma unroll
        for (int r = 0; r < 8; ++r) {
            const int m = (l < 16) ? r : r + 8;
            const float x0 = xin[(p + m) * CH + m0];
            const float x1 = xin[(p + m) * CH + 16 + m0];
            xout[(p + m) * CH + m0]      = gelu_f(o0[r] + bslo + x0);
            xout[(p + m) * CH + 16 + m0] = gelu_f(o1[r] + bshi + x1);
        }
    }
}

// ---------------------------------------------------------------------------
// Projection head: out = gelu(x*W1 + b1) . w2 + b2, with z-crop 46->40.
// All 8 B fragments of W1 (32x128) hoisted; T_Q tiles per wave.
__global__ __launch_bounds__(128) void k_q(const float* __restrict__ xin,
                                           const _Float16* __restrict__ W1,
                                           const float* __restrict__ b1,
                                           const float* __restrict__ w2,
                                           const float* __restrict__ b2,
                                           float* __restrict__ out) {
    const int l = threadIdx.x & 31;
    const int w = threadIdx.x >> 5;
    const int tile0 = (blockIdx.x * 4 + w) * T_Q;
    const int m0 = l & 15;

    v16h Bq[8];
    float b1n[8], w2n[8];
#pragma unroll
    for (int nt = 0; nt < 8; ++nt) {
        Bq[nt]  = load_b_f16(W1, nt, l);
        b1n[nt] = b1[nt * 16 + m0];
        w2n[nt] = w2[nt * 16 + m0];
    }
    const float b2v = b2[0];

    for (int t = 0; t < T_Q; ++t) {
        const int tile = tile0 + t;
        const int p2 = tile * 16 + m0;          // cropped pixel index
        const int col = p2 / NZ0;
        const int z = p2 - col * NZ0;
        v16h ax = load_a_f32(xin + (col * NZ + z) * CH, l);
        float oacc[8];
#pragma unroll
        for (int r = 0; r < 8; ++r) oacc[r] = 0.0f;
#pragma unroll
        for (int nt = 0; nt < 8; ++nt) {
            v8f acc = {};
            acc = wmma32(ax, Bq[nt], acc);
#pragma unroll
            for (int r = 0; r < 8; ++r) oacc[r] += gelu_f(acc[r] + b1n[nt]) * w2n[nt];
        }
#pragma unroll
        for (int r = 0; r < 8; ++r) {
            float v = oacc[r];
            v += __shfl_xor(v, 1, 32);
            v += __shfl_xor(v, 2, 32);
            v += __shfl_xor(v, 4, 32);
            v += __shfl_xor(v, 8, 32);
            if (m0 == 0) {
                const int m = (l < 16) ? r : r + 8;
                out[tile * 16 + m] = v + b2v;
            }
        }
    }
}

// ---------------------------------------------------------------------------
extern "C" void kernel_launch(void* const* d_in, const int* in_sizes, int n_in,
                              void* d_out, int out_size, void* d_ws, size_t ws_size,
                              hipStream_t stream) {
    // Locate x by its unique size; params leaves assumed in jax tree_flatten
    // (sorted-key) order either after x (x first) or before it (x last).
    const int XSZ = BATCH * NX * NY * NZ0 * 7;  // 4587520
    int xi = n_in - 1;
    for (int i = 0; i < n_in; ++i)
        if (in_sizes[i] == XSZ) { xi = i; break; }
    const int pbase = (xi == 0) ? 1 : 0;
    auto P = [&](int k) -> const float* { return (const float*)d_in[pbase + k]; };

    const float* Xin = (const float*)d_in[xi];
    const float* convW[4][4];
    for (int i = 0; i < 4; ++i)
        for (int j = 0; j < 4; ++j) convW[i][j] = P(i * 4 + j);
    const float *mlpB1[4], *mlpB2[4], *mlpW1[4], *mlpW2[4];
    for (int i = 0; i < 4; ++i) {
        mlpB1[i] = P(16 + i * 4 + 0);
        mlpB2[i] = P(16 + i * 4 + 1);
        mlpW1[i] = P(16 + i * 4 + 2);
        mlpW2[i] = P(16 + i * 4 + 3);
    }
    const float* p_b  = P(32);
    const float* p_w  = P(33);
    const float* q_b1 = P(34);
    const float* q_b2 = P(35);
    const float* q_w1 = P(36);
    const float* q_w2 = P(37);
    const float *wB[4], *wW[4];
    for (int i = 0; i < 4; ++i) { wB[i] = P(38 + i * 2); wW[i] = P(38 + i * 2 + 1); }

    // Workspace carve-up (~239 MB).
    const size_t nBig = (size_t)NPIX * CH;                       // 24,117,248
    const size_t nSz  = (size_t)NCOL * MK * 2 * CH;              //  8,388,608
    const size_t nSzy = (size_t)BATCH * NX * MXY * MK * 2 * CH;  //  2,097,152
    const size_t nSm  = (size_t)BATCH * MXY * MXY * MK * 2 * CH; //    524,288
    const size_t need = (2 * nBig + nSz + nSzy + 2 * nSm) * 4 + 16384 * 2;
    if (ws_size < need) return;

    float* bufA = (float*)d_ws;
    float* bufB = bufA + nBig;
    float* Sz   = bufB + nBig;
    float* Szy  = Sz + nSz;
    float* Sm   = Szy + nSzy;
    float* Ym   = Sm + nSm;
    _Float16* fw = (_Float16*)(Ym + nSm);
    // fw: per block i: [mlpW1 1024][mlpW2 1024][skipW 1024], then qW1 4096 f16.
    auto fMlpW1 = [&](int i) { return fw + i * 3072; };
    auto fMlpW2 = [&](int i) { return fw + i * 3072 + 1024; };
    auto fSkipW = [&](int i) { return fw + i * 3072 + 2048; };
    _Float16* fQ1 = fw + 12288;

    for (int i = 0; i < 4; ++i) {
        k_cvt<<<4, 256, 0, stream>>>(mlpW1[i], fMlpW1(i), 1024);
        k_cvt<<<4, 256, 0, stream>>>(mlpW2[i], fMlpW2(i), 1024);
        k_cvt<<<4, 256, 0, stream>>>(wW[i],    fSkipW(i), 1024);
    }
    k_cvt<<<16, 256, 0, stream>>>(q_w1, fQ1, 4096);

    k_lift<<<NCOL, 256, 0, stream>>>(Xin, p_w, p_b, bufA);

    float* cur = bufA;
    float* alt = bufB;
    const int pwBlocks = (NPIX / 16) / (4 * T_PW);   // 2944
    for (int i = 0; i < 4; ++i) {
        if (i == 2) {                            // spectral branch is dead for block 2
            k_skip_only<<<pwBlocks, 128, 0, stream>>>(cur, fSkipW(2), wB[2], alt);
            float* t = cur; cur = alt; alt = t;
            continue;
        }
        k_dft_z<<<NCOL / 8, 256, 0, stream>>>(cur, Sz);
        k_dft_y<<<BATCH * NX * MXY, 256, 0, stream>>>(Sz, Szy);
        k_dft_x<<<BATCH * MXY * MXY, 256, 0, stream>>>(Szy, Sm);
        k_modemul<<<BATCH * MXY * MXY, 256, 0, stream>>>(
            Sm, convW[i][0], convW[i][1], convW[i][2], convW[i][3], Ym);
        k_idft_x<<<BATCH * NX * MXY, 256, 0, stream>>>(Ym, Szy);
        k_idft_y<<<NCOL, 256, 0, stream>>>(Szy, Sz);
        k_idft_z<<<NCOL, 256, 0, stream>>>(Sz, alt);    // alt <- x1 spatial
        const int act = (i < 3) ? 1 : 2;
        k_block_pw<<<pwBlocks, 128, 0, stream>>>(alt, cur, fMlpW1(i), mlpB1[i],
                                                 fMlpW2(i), mlpB2[i], fSkipW(i),
                                                 wB[i], act);
        float* t = cur; cur = alt; alt = t;
    }

    k_q<<<(NPIX0 / 16) / (4 * T_Q), 128, 0, stream>>>(cur, fQ1, q_b1, q_w2, q_b2,
                                                      (float*)d_out);
}
